// AttnRnnModel_19688130085637
// MI455X (gfx1250) — matrix operand order repaired and checked by reading, not compiled, not run on hardware
//
#include <hip/hip_runtime.h>
#include <stdint.h>
#include <math.h>

// ---------------- CDNA5 WMMA types ----------------
typedef __attribute__((ext_vector_type(16))) _Float16 v16h;
typedef __attribute__((ext_vector_type(8)))  _Float16 v8h;
typedef __attribute__((ext_vector_type(8)))  float    v8f;
typedef __attribute__((ext_vector_type(4)))  unsigned int v4u;
typedef __attribute__((ext_vector_type(8)))  int      v8i;
typedef __attribute__((ext_vector_type(4)))  int      v4i;

#define B_     64
#define T_     128
#define HID_   512
#define DIN_   320      // LOC_EMB + TIM_EMB
#define DH3_   1536     // 3*HID
#define NG_    256      // history groups (1024/4)
#define DATT_  448      // LOC_EMB + TIM_EMB(64) + UID_EMB(128)
#define NUID_  5000
#define NUIDP_ 5008     // padded to multiple of 16

union V16U { v16h v; v8h h[2]; };

// A-tile (16x32 f16, row-major, leading dim ld), ISA 7.12.2 layout:
// lanes 0-15: M=lane, K 0..7 + 16..23 ; lanes 16-31: M=lane-16, K 8..15 + 24..31
__device__ __forceinline__ v16h load_a16(const _Float16* base, int ld) {
  int lane = threadIdx.x & 31;
  int row  = lane & 15;
  int kof  = (lane >> 4) << 3;          // 0 or 8
  const _Float16* p = base + row * ld + kof;
  V16U u;
  u.h[0] = *reinterpret_cast<const v8h*>(p);
  u.h[1] = *reinterpret_cast<const v8h*>(p + 16);
  return u.v;
}

// B-tile for C = X * W^T, W row-major (N,K): column n of B is row n of W.
// lanes 0-15: N=lane, K 0..7 + 16..23 ; lanes 16-31: N=lane-16, K 8..15 + 24..31
__device__ __forceinline__ v16h load_b16(const _Float16* wrow, int ld) {
  int lane = threadIdx.x & 31;
  int col  = lane & 15;
  int kof  = (lane >> 4) << 3;          // 0 or 8
  const _Float16* p = wrow + col * ld + kof;
  V16U u;
  u.h[0] = *reinterpret_cast<const v8h*>(p);
  u.h[1] = *reinterpret_cast<const v8h*>(p + 16);
  return u.v;
}

__device__ __forceinline__ v8f wmma16(v16h a, v16h b, v8f c) {
  return __builtin_amdgcn_wmma_f32_16x16x32_f16(false, a, false, b, (short)0, c,
                                                false, false);
}

// ---------------- Tensor Data Mover: 1-D contiguous block -> LDS ----------
#if __has_builtin(__builtin_amdgcn_tensor_load_to_lds) && __has_builtin(__builtin_amdgcn_s_wait_tensorcnt)
#define HAS_TDM_ 1
// Build D# per CDNA5 ISA 8.3/8.4: group0 = {count/lds_addr/global_addr/type},
// group1 = {data_size=4B, tensor_dim0 = tile_dim0 = ndw dwords, tile_dim1 = 1}.
__device__ __forceinline__ void tdm_load_1d(unsigned int lds_off, uint64_t gaddr,
                                            unsigned int ndw) {
  v4u g0;
  g0[0] = 1u;                                              // count=1 (valid), user mode
  g0[1] = lds_off;                                         // lds_addr (bytes)
  g0[2] = (unsigned int)(gaddr & 0xFFFFFFFFull);           // global_addr[31:0]
  g0[3] = (unsigned int)((gaddr >> 32) & 0x01FFFFFFull)    // global_addr[56:32]
          | (2u << 30);                                    // type = 2 ("image")
  v8i g1;
  g1[0] = (int)(2u << 16);                                 // data_size=2 -> 4-byte elems
  g1[1] = (int)((ndw & 0xFFFFu) << 16);                    // tensor_dim0[15:0]
  g1[2] = (int)(((ndw >> 16) & 0xFFFFu) | (1u << 16));     // tensor_dim0[31:16], tensor_dim1=1
  g1[3] = (int)((ndw & 0xFFFFu) << 16);                    // tile_dim0 = ndw
  g1[4] = 1;                                               // tile_dim1 = 1, tile_dim2 = 0
  g1[5] = (int)ndw;                                        // tensor_dim0_stride[31:0]
  g1[6] = 0;
  g1[7] = 0;
  v4i z = {0, 0, 0, 0};
#if __has_include(<hip/amd_detail/amd_gfx1250_TDM.h>)
  v8i z8 = {0, 0, 0, 0, 0, 0, 0, 0};
  __builtin_amdgcn_tensor_load_to_lds(g0, g1, z, z, z8, 0);   // 6-arg toolchain
#else
  __builtin_amdgcn_tensor_load_to_lds(g0, g1, z, z, 0);       // 5-arg (ROCm 7.2)
#endif
}
#endif

// Stage a contiguous block of n_halves f16 (multiple of 16) into LDS, then barrier.
__device__ __forceinline__ void stage_block(_Float16* lds, const _Float16* g,
                                            int n_halves) {
#ifdef HAS_TDM_
  if (threadIdx.x == 0) {
    tdm_load_1d((unsigned int)(uintptr_t)lds, (uint64_t)(uintptr_t)g,
                (unsigned int)(n_halves >> 1));
    __builtin_amdgcn_s_wait_tensorcnt(0);
  }
#else
  for (int i = threadIdx.x; i < (n_halves >> 3); i += blockDim.x)
    reinterpret_cast<v8h*>(lds)[i] = reinterpret_cast<const v8h*>(g)[i];
#endif
  __syncthreads();
}

// ---------------- precision conversion ----------------
__global__ void cvt_f32_f16(const float* __restrict__ in, _Float16* __restrict__ out, int n) {
  int i = blockIdx.x * 256 + threadIdx.x;
  if (i < n) out[i] = (_Float16)in[i];
}

__global__ void cvt_pad_final(const float* __restrict__ in, _Float16* __restrict__ out) {
  long i = (long)blockIdx.x * 256 + threadIdx.x;
  long total = (long)NUIDP_ * DH3_;
  if (i >= total) return;
  long row = i / DH3_, col = i - row * DH3_;
  out[i] = (row < NUID_) ? (_Float16)in[row * DH3_ + col] : (_Float16)0.f;
}

// ---------------- embedding gather: x = [emb_loc | emb_tim] ----------------
__global__ void gather_x(const int* __restrict__ loc, const int* __restrict__ tim,
                         const float* __restrict__ eloc, const float* __restrict__ etim,
                         _Float16* __restrict__ xh) {
  int row = blockIdx.x;                 // b*T + t
  int l = loc[row], tt = tim[row];
  _Float16* o = xh + (size_t)row * DIN_;
  for (int c = threadIdx.x; c < 256; c += 256) o[c] = (_Float16)eloc[(size_t)l * 256 + c];
  for (int c = threadIdx.x; c < 64;  c += 256) o[256 + c] = (_Float16)etim[tt * 64 + c];
}

// ---------------- gx = x @ W_ih^T + b_ih, stored (T,B,3H) f32 --------------
// Block = 4 waves sharing one 16-row A block (TDM-staged in LDS).
__global__ void __launch_bounds__(128) gemm_gx(const _Float16* __restrict__ xh,
                                               const _Float16* __restrict__ Wih,
                                               const float* __restrict__ bih,
                                               float* __restrict__ gx) {
  __shared__ _Float16 As[16 * DIN_];    // 10 KB
  int wave = threadIdx.x >> 5;
  int tm = blockIdx.x / 24;             // 96 n-tiles / 4 waves
  int tn = (blockIdx.x % 24) * 4 + wave;
  int m0 = tm * 16, n0 = tn * 16;
  stage_block(As, xh + (size_t)m0 * DIN_, 16 * DIN_);
  v8f c = {};
  for (int k = 0; k < DIN_; k += 32) {
    v16h a = load_a16(As + k, DIN_);                           // LDS
    v16h b = load_b16(Wih + (size_t)n0 * DIN_ + k, DIN_);      // L2
    c = wmma16(a, b, c);
  }
  int lane = threadIdx.x & 31;
  int col = lane & 15, hi = (lane >> 4) << 3;
  int bb = m0 >> 7;                     // batch (T=128, tiles never straddle)
  int t0 = m0 & 127;
  int n = n0 + col;
  float bias = bih[n];
#pragma unroll
  for (int r = 0; r < 8; ++r) {
    int t = t0 + hi + r;
    gx[((size_t)t * B_ + bb) * DH3_ + n] = c[r] + bias;
  }
}

// ---------------- GRU recurrence: one WGP, h double-buffered in LDS --------
__global__ void __launch_bounds__(1024) gru_kernel(const float* __restrict__ gx,
                                                   const _Float16* __restrict__ Whh,
                                                   const float* __restrict__ bhh,
                                                   const int* __restrict__ lens,
                                                   float* __restrict__ last) {
  extern __shared__ _Float16 smem[];    // 2 * 64 * 512 f16 = 128 KB
  _Float16 (*hb)[B_][HID_] = (_Float16 (*)[B_][HID_])smem;
  for (int i = threadIdx.x; i < 2 * B_ * HID_; i += 1024) smem[i] = (_Float16)0.f;
  __syncthreads();

  int wave = threadIdx.x >> 5;
  int lane = threadIdx.x & 31;
  int col = lane & 15, hi = (lane >> 4) << 3;

  for (int t = 0; t < T_; ++t) {
    const _Float16 (*h)[HID_] = hb[t & 1];
    _Float16 (*hn)[HID_] = hb[(t + 1) & 1];
    const float* gxt = gx + (size_t)t * B_ * DH3_;
    // prefetch next step's gx slice (384 KB) while this step computes
    if (t + 1 < T_) {
      const char* nx = (const char*)(gx + (size_t)(t + 1) * B_ * DH3_);
      for (size_t off = (size_t)threadIdx.x * 128; off < (size_t)B_ * DH3_ * 4;
           off += 1024u * 128u)
        __builtin_prefetch(nx + off, 0, 1);
    }
    for (int q = 0; q < 4; ++q) {
      int g = wave * 4 + q;             // 0..127 gate-tile groups
      int m0 = (g >> 5) << 4;           // batch tile (4)
      int n0 = (g & 31) << 4;           // hidden tile (32)
      v8f cr = {}, cz = {}, cn = {};
      for (int k = 0; k < HID_; k += 32) {
        v16h a = load_a16(&h[m0][k], HID_);
        cr = wmma16(a, load_b16(Whh + (size_t)(n0)            * HID_ + k, HID_), cr);
        cz = wmma16(a, load_b16(Whh + (size_t)(HID_ + n0)     * HID_ + k, HID_), cz);
        cn = wmma16(a, load_b16(Whh + (size_t)(2 * HID_ + n0) * HID_ + k, HID_), cn);
      }
      int n = n0 + col;
      float br = bhh[n], bz = bhh[HID_ + n], bn = bhh[2 * HID_ + n];
#pragma unroll
      for (int r = 0; r < 8; ++r) {
        int m = m0 + hi + r;            // batch index
        const float* gxr = gxt + (size_t)m * DH3_;
        float rr  = 1.f / (1.f + __expf(-(gxr[n] + cr[r] + br)));
        float zz  = 1.f / (1.f + __expf(-(gxr[HID_ + n] + cz[r] + bz)));
        float nnv = tanhf(gxr[2 * HID_ + n] + rr * (cn[r] + bn));
        float hold = (float)h[m][n];
        float hnew = (1.f - zz) * nnv + zz * hold;
        hn[m][n] = (_Float16)hnew;
        if (lens[m] - 1 == t) last[(size_t)m * HID_ + n] = hnew;
      }
    }
    __syncthreads();
  }
}

// ---------------- history gather: [loc | tim | mean(uid group)] ------------
__global__ void gather_hist(const int* __restrict__ hloc, const int* __restrict__ htim,
                            const int* __restrict__ huid,
                            const float* __restrict__ eloc, const float* __restrict__ etim,
                            const float* __restrict__ euid,
                            _Float16* __restrict__ hin) {
  int row = blockIdx.x;                 // b*256 + g
  int b = row >> 8, g = row & 255;
  int base = b * 1024 + g * 4;          // L_HIST=1024, G=4
  int l = hloc[base], tt = htim[base];
  int u0 = huid[base], u1 = huid[base + 1], u2 = huid[base + 2], u3 = huid[base + 3];
  _Float16* o = hin + (size_t)row * DATT_;
  for (int c = threadIdx.x; c < 256; c += 128) o[c] = (_Float16)eloc[(size_t)l * 256 + c];
  for (int c = threadIdx.x; c < 64;  c += 128) o[256 + c] = (_Float16)etim[tt * 64 + c];
  for (int c = threadIdx.x; c < 128; c += 128)
    o[320 + c] = (_Float16)(0.25f * (euid[u0 * 128 + c] + euid[u1 * 128 + c] +
                                     euid[u2 * 128 + c] + euid[u3 * 128 + c]));
}

// ---------------- hist = tanh(hist_in @ W_attn^T + b), f32 (B,256,512) -----
__global__ void __launch_bounds__(128) gemm_hist(const _Float16* __restrict__ hin,
                                                 const _Float16* __restrict__ Wat,
                                                 const float* __restrict__ bat,
                                                 float* __restrict__ hist) {
  __shared__ _Float16 As[16 * DATT_];   // 14 KB
  int wave = threadIdx.x >> 5;
  int tm = blockIdx.x / 8;              // 32 n-tiles / 4 waves
  int tn = (blockIdx.x % 8) * 4 + wave;
  int m0 = tm * 16, n0 = tn * 16;
  stage_block(As, hin + (size_t)m0 * DATT_, 16 * DATT_);
  v8f c = {};
  for (int k = 0; k < DATT_; k += 32) {
    v16h a = load_a16(As + k, DATT_);
    v16h b = load_b16(Wat + (size_t)n0 * DATT_ + k, DATT_);
    c = wmma16(a, b, c);
  }
  int lane = threadIdx.x & 31;
  int col = lane & 15, hi = (lane >> 4) << 3;
  int n = n0 + col;
  float bias = bat[n];
#pragma unroll
  for (int r = 0; r < 8; ++r) {
    int m = m0 + hi + r;
    hist[(size_t)m * HID_ + n] = tanhf(c[r] + bias);
  }
}

// ---------------- energies + softmax + context, per batch row --------------
__global__ void __launch_bounds__(256) attn_kernel(const float* __restrict__ last,
                                                   const float* __restrict__ hist,
                                                   float* __restrict__ context) {
  int b = blockIdx.x;
  int g = threadIdx.x;                  // 0..255 group
  const float* hb = hist + (size_t)b * NG_ * HID_;
  const float* lb = last + (size_t)b * HID_;
  const float* hg = hb + (size_t)g * HID_;
  float e = 0.f;
  for (int h = 0; h < HID_; ++h) e += lb[h] * hg[h];
  __shared__ float red[256];
  __shared__ float w[256];
  red[g] = e; __syncthreads();
  for (int s = 128; s > 0; s >>= 1) { if (g < s) red[g] = fmaxf(red[g], red[g + s]); __syncthreads(); }
  float mx = red[0]; __syncthreads();
  float ex = __expf(e - mx);
  red[g] = ex; __syncthreads();
  for (int s = 128; s > 0; s >>= 1) { if (g < s) red[g] += red[g + s]; __syncthreads(); }
  float inv = 1.f / red[0];
  w[g] = ex * inv;
  __syncthreads();
  for (int h = g; h < HID_; h += 256) {
    float acc = 0.f;
    for (int gg = 0; gg < NG_; ++gg) acc += w[gg] * hb[(size_t)gg * HID_ + h];
    context[(size_t)b * HID_ + h] = acc;
  }
}

// ---------------- out_cat = f16 concat(last, context, last) ----------------
__global__ void build_cat(const float* __restrict__ last, const float* __restrict__ context,
                          _Float16* __restrict__ cat) {
  int b = blockIdx.x;
  for (int c = threadIdx.x; c < HID_; c += 256) {
    _Float16 l = (_Float16)last[(size_t)b * HID_ + c];
    cat[(size_t)b * DH3_ + c] = l;
    cat[(size_t)b * DH3_ + HID_ + c] = (_Float16)context[(size_t)b * HID_ + c];
    cat[(size_t)b * DH3_ + 2 * HID_ + c] = l;
  }
}

// ---------------- y = out_cat @ W_final^T + b_final ------------------------
// Block = one n-tile; its 4 waves take the 4 m-tiles and share the 16-row
// W_final block (48 KB), TDM-staged into LDS.
__global__ void __launch_bounds__(128) gemm_final(const _Float16* __restrict__ cat,
                                                  const _Float16* __restrict__ Wf,
                                                  const float* __restrict__ bf,
                                                  float* __restrict__ y) {
  __shared__ _Float16 Bs[16 * DH3_];    // 48 KB
  int tm = threadIdx.x >> 5;            // wave -> m-tile (M=64 -> 4 tiles)
  int n0 = blockIdx.x * 16;
  int m0 = tm * 16;
  stage_block(Bs, Wf + (size_t)n0 * DH3_, 16 * DH3_);
  v8f c = {};
  for (int k = 0; k < DH3_; k += 32) {
    v16h a = load_a16(cat + (size_t)m0 * DH3_ + k, DH3_);  // L2 (tiny)
    v16h b = load_b16(Bs + k, DH3_);                       // LDS
    c = wmma16(a, b, c);
  }
  int lane = threadIdx.x & 31;
  int col = lane & 15, hi = (lane >> 4) << 3;
  int n = n0 + col;
  if (n < NUID_) {
    float bias = bf[n];
#pragma unroll
    for (int r = 0; r < 8; ++r) {
      int m = m0 + hi + r;
      y[(size_t)m * NUID_ + n] = c[r] + bias;
    }
  }
}

// ---------------- broadcast (B,5000) -> (B,T,5000) -------------------------
__global__ void bcast_out(const float* __restrict__ y, float* __restrict__ out) {
  int bt = blockIdx.x;                  // b*T + t
  int b = bt / T_;
  const float* yb = y + (size_t)b * NUID_;
  float* ob = out + (size_t)bt * NUID_;
  for (int j = threadIdx.x; j < NUID_; j += 256) ob[j] = yb[j];
}

// ===========================================================================
extern "C" void kernel_launch(void* const* d_in, const int* in_sizes, int n_in,
                              void* d_out, int out_size, void* d_ws, size_t ws_size,
                              hipStream_t stream) {
  const int*   loc     = (const int*)d_in[0];
  const int*   tim     = (const int*)d_in[1];
  const int*   lens    = (const int*)d_in[2];
  const int*   hloc    = (const int*)d_in[3];
  const int*   htim    = (const int*)d_in[4];
  const int*   huid    = (const int*)d_in[5];
  /* d_in[6] = group_size (==4, baked in) */
  const float* emb_loc = (const float*)d_in[7];
  const float* emb_tim = (const float*)d_in[8];
  const float* emb_uid = (const float*)d_in[9];
  const float* W_attn  = (const float*)d_in[10];
  const float* b_attn  = (const float*)d_in[11];
  const float* W_ih    = (const float*)d_in[12];
  const float* b_ih    = (const float*)d_in[13];
  const float* W_hh    = (const float*)d_in[14];
  const float* b_hh    = (const float*)d_in[15];
  const float* W_final = (const float*)d_in[16];
  const float* b_final = (const float*)d_in[17];

  char* ws = (char*)d_ws;
  size_t off = 0;
  auto alloc = [&](size_t bytes) { char* p = ws + off; off += (bytes + 255) & ~(size_t)255; return p; };

  _Float16* Wih_h  = (_Float16*)alloc((size_t)DH3_ * DIN_ * 2);
  _Float16* Whh_h  = (_Float16*)alloc((size_t)DH3_ * HID_ * 2);
  _Float16* Wat_h  = (_Float16*)alloc((size_t)HID_ * DATT_ * 2);
  _Float16* Wf_h   = (_Float16*)alloc((size_t)NUIDP_ * DH3_ * 2);
  _Float16* xh     = (_Float16*)alloc((size_t)B_ * T_ * DIN_ * 2);
  float*    gx     = (float*)   alloc((size_t)T_ * B_ * DH3_ * 4);
  float*    lastb  = (float*)   alloc((size_t)B_ * HID_ * 4);
  _Float16* hin    = (_Float16*)alloc((size_t)B_ * NG_ * DATT_ * 2);
  float*    hist   = (float*)   alloc((size_t)B_ * NG_ * HID_ * 4);
  float*    ctx    = (float*)   alloc((size_t)B_ * HID_ * 4);
  _Float16* cat    = (_Float16*)alloc((size_t)B_ * DH3_ * 2);
  float*    y      = (float*)   alloc((size_t)B_ * NUID_ * 4);
  (void)ws_size; (void)n_in; (void)in_sizes; (void)out_size;

  // weight conversion to f16
  cvt_f32_f16<<<(DH3_ * DIN_ + 255) / 256, 256, 0, stream>>>(W_ih, Wih_h, DH3_ * DIN_);
  cvt_f32_f16<<<(DH3_ * HID_ + 255) / 256, 256, 0, stream>>>(W_hh, Whh_h, DH3_ * HID_);
  cvt_f32_f16<<<(HID_ * DATT_ + 255) / 256, 256, 0, stream>>>(W_attn, Wat_h, HID_ * DATT_);
  cvt_pad_final<<<(int)(((size_t)NUIDP_ * DH3_ + 255) / 256), 256, 0, stream>>>(W_final, Wf_h);

  // x gather + gx GEMM (M=8192, N=1536, K=320)
  gather_x<<<B_ * T_, 256, 0, stream>>>(loc, tim, emb_loc, emb_tim, xh);
  gemm_gx<<<(B_ * T_ / 16) * (DH3_ / 16) / 4, 128, 0, stream>>>(xh, Wih_h, b_ih, gx);

  // GRU: single 1024-thread workgroup, 128 KB dynamic LDS (double-buffered h)
  gru_kernel<<<1, 1024, 2 * B_ * HID_ * sizeof(_Float16), stream>>>(gx, Whh_h, b_hh, lens, lastb);

  // history attention
  gather_hist<<<B_ * NG_, 128, 0, stream>>>(hloc, htim, huid, emb_loc, emb_tim, emb_uid, hin);
  gemm_hist<<<(B_ * NG_ / 16) * (HID_ / 16) / 4, 128, 0, stream>>>(hin, Wat_h, b_attn, hist);
  attn_kernel<<<B_, 256, 0, stream>>>(lastb, hist, ctx);

  // final projection + broadcast
  build_cat<<<B_, 256, 0, stream>>>(lastb, ctx, cat);
  gemm_final<<<NUIDP_ / 16, 128, 0, stream>>>(cat, Wf_h, b_final, y);
  bcast_out<<<B_ * T_, 256, 0, stream>>>(y, (float*)d_out);
}
